// GNN_67774583931071
// MI455X (gfx1250) — compile-verified, matrix operand order checked by reading, not run on hardware
//
#include <hip/hip_runtime.h>

typedef __attribute__((ext_vector_type(16))) __bf16 v16bf;
typedef __attribute__((ext_vector_type(8)))  float  v8f;

// ---------------- degree / normalization ----------------

__global__ void deg_init_kernel(float* deg, int n) {
  int i = blockIdx.x * blockDim.x + threadIdx.x;
  if (i < n) deg[i] = 1.0f;                     // self-loop contribution
}

__global__ void deg_count_kernel(const int* __restrict__ dst, float* deg, int E) {
  int e = blockIdx.x * blockDim.x + threadIdx.x;
  if (e < E) atomicAdd(&deg[dst[e]], 1.0f);
}

__global__ void deg_rsqrt_kernel(float* deg, int n) {
  int i = blockIdx.x * blockDim.x + threadIdx.x;
  if (i < n) deg[i] = rsqrtf(deg[i]);           // dinv = deg^-1/2
}

// ---------------- strip GEMM via WMMA (bf16 in, f32 accum) ----------------
// Each wave32 owns a 16-row strip of H = A(N x DIN) * W(DIN x DOUT):
//  - A fragments for the strip loaded once into VGPRs (reused across all n-tiles)
//  - W staged bf16-swizzled in LDS per block (two column-halves <= 64KB each),
//    so a lane's 16-element B fragment is one contiguous 32B chunk (ds_load_b128 x2)
// H must have capacity for ceil(n_rows/16)*16 rows (workspace is padded), so
// stores are unguarded -> no exec-mask branching around the WMMA pipeline.

template <int DIN, int DOUT>
__global__ __launch_bounds__(256)
void gcn_gemm_wmma_kernel(const float* __restrict__ A, const float* __restrict__ W,
                          float* __restrict__ H, int n_rows) {
  constexpr int TK  = DIN / 32;   // k-tiles of 32
  constexpr int TN  = DOUT / 16;  // n-tiles of 16
  constexpr int TNH = TN / 2;     // n-tiles per LDS half
  __shared__ __align__(32) __bf16 wlds[TK * TNH * 32 * 16];  // <= 64KB

  const int lane = threadIdx.x & 31;
  const int wid  = threadIdx.x >> 5;
  const int tiles_m = (n_rows + 15) >> 4;
  int tm = blockIdx.x * 8 + wid;
  const bool active = (tm < tiles_m);           // wave-uniform
  if (!active) tm = tiles_m - 1;                // idle waves shadow last strip

  const int l15  = lane & 15;
  const int kgrp = (lane >> 4) << 3;            // 0 or 8 (== 8*lane[4])

  int m = tm * 16 + l15;
  if (m >= n_rows) m = n_rows - 1;              // clamp loads, EXEC stays full

  // A fragments for the whole strip, kept in VGPRs.
  v16bf afrag[TK];
  const float* arow = A + (size_t)m * DIN + kgrp;
#pragma unroll
  for (int kk = 0; kk < TK; ++kk) {
#pragma unroll
    for (int i = 0; i < 8; ++i) {
      afrag[kk][i]     = (__bf16)arow[kk * 32 + i];        // K = 32kk+kgrp+i
      afrag[kk][i + 8] = (__bf16)arow[kk * 32 + 16 + i];   // K = 32kk+16+kgrp+i
    }
  }

  for (int nh = 0; nh < 2; ++nh) {
    __syncthreads();                            // protect LDS reuse
    constexpr int NCOLS = DOUT / 2;
    for (int idx = threadIdx.x; idx < DIN * NCOLS; idx += 256) {
      int k  = idx / NCOLS;                     // NCOLS is a power of two
      int n  = idx - k * NCOLS;                 // column within this half
      int kk = k >> 5, kr = k & 31;
      int lanehi = (kr >> 3) & 1;
      int e      = (kr & 7) | ((kr >> 4) << 3);
      int slane  = (lanehi << 4) | (n & 15);
      int tn     = n >> 4;
      wlds[((tn * TK + kk) * 32 + slane) * 16 + e] =
          (__bf16)W[(size_t)k * DOUT + (nh * NCOLS + n)];
    }
    __syncthreads();

    for (int tn = 0; tn < TNH; ++tn) {
      v8f acc = {};
#pragma unroll
      for (int kk = 0; kk < TK; ++kk) {
        v16bf b = *(const v16bf*)&wlds[((tn * TK + kk) * 32 + lane) * 16];
        acc = __builtin_amdgcn_wmma_f32_16x16x32_bf16(
            /*neg_a=*/false, afrag[kk], /*neg_b=*/false, b,
            /*c_mod=*/(short)0, acc, /*reuse_a=*/false, /*reuse_b=*/false);
      }
      if (active) {                             // wave-uniform; no lane masking
        const int gtn   = nh * TNH + tn;
        const int mbase = tm * 16 + kgrp;
        float* out = H + (size_t)mbase * DOUT + gtn * 16 + l15;
#pragma unroll
        for (int r = 0; r < 8; ++r) out[(size_t)r * DOUT] = acc[r];
      }
    }
  }
}

// ---------------- out = h*dinv^2 + bias (self loop + bias) ----------------

__global__ void self_bias_kernel(const float* __restrict__ h, const float* __restrict__ dinv,
                                 const float* __restrict__ bias, float* __restrict__ out,
                                 int n, int d) {
  size_t idx = (size_t)blockIdx.x * blockDim.x + threadIdx.x;
  if (idx >= (size_t)n * d) return;
  int i = (int)(idx / d);
  int j = (int)(idx - (size_t)i * d);
  float s = dinv[i];
  out[idx] = h[idx] * (s * s) + bias[j];
}

// ---------------- edge gather-scale-scatter (atomic) ----------------

__global__ void edge_scatter_kernel(const float* __restrict__ h, const int* __restrict__ src,
                                    const int* __restrict__ dst, const float* __restrict__ dinv,
                                    float* out, int d) {
  int e = blockIdx.x;
  int j = threadIdx.x;
  int s = src[e], t = dst[e];
  float w = dinv[s] * dinv[t];
  atomicAdd(&out[(size_t)t * d + j], h[(size_t)s * d + j] * w);
}

__global__ void relu_kernel(float* x, size_t n) {
  size_t i = (size_t)blockIdx.x * blockDim.x + threadIdx.x;
  if (i < n) x[i] = fmaxf(x[i], 0.0f);
}

// ---------------- driver ----------------

extern "C" void kernel_launch(void* const* d_in, const int* in_sizes, int n_in,
                              void* d_out, int out_size, void* d_ws, size_t ws_size,
                              hipStream_t stream) {
  const float* x  = (const float*)d_in[0];
  const int*   ei = (const int*)d_in[1];
  const float* W1 = (const float*)d_in[2];
  const float* b1 = (const float*)d_in[3];
  const float* W2 = (const float*)d_in[4];
  const float* b2 = (const float*)d_in[5];
  const float* W3 = (const float*)d_in[6];
  const float* b3 = (const float*)d_in[7];

  const int D0 = 128;
  const int N = in_sizes[0] / D0;
  const int E = in_sizes[1] / 2;
  const int* srcIdx = ei;       // edge_index[0]
  const int* dstIdx = ei + E;   // edge_index[1]

  const size_t Npad = ((size_t)N + 15) & ~(size_t)15;       // GEMM stores full tiles
  float* dinv = (float*)d_ws;
  float* hA   = dinv + (((size_t)N + 255) & ~(size_t)255);  // Npad x 256 scratch (XW)
  float* hB   = hA + Npad * 256;                             // Npad x 256 scratch (layer out)
  float* outF = (float*)d_out;

  // normalization: deg = 1 + indegree; dinv = rsqrt(deg)
  deg_init_kernel <<<(N + 255) / 256, 256, 0, stream>>>(dinv, N);
  deg_count_kernel<<<(E + 255) / 256, 256, 0, stream>>>(dstIdx, dinv, E);
  deg_rsqrt_kernel<<<(N + 255) / 256, 256, 0, stream>>>(dinv, N);

  const int tiles_m     = (N + 15) / 16;
  const int gemm_blocks = (tiles_m + 7) / 8;

  auto post = [&](const float* h, const float* b, float* out, int dd) {
    size_t tot = (size_t)N * dd;
    self_bias_kernel<<<(int)((tot + 255) / 256), 256, 0, stream>>>(h, dinv, b, out, N, dd);
    edge_scatter_kernel<<<E, dd, 0, stream>>>(h, srcIdx, dstIdx, dinv, out, dd);
    relu_kernel<<<(int)((tot + 255) / 256), 256, 0, stream>>>(out, tot);
  };

  // Layer 1: 128 -> 256
  gcn_gemm_wmma_kernel<128, 256><<<gemm_blocks, 256, 0, stream>>>(x, W1, hA, N);
  post(hA, b1, hB, 256);
  // Layer 2: 256 -> 256  (hB consumed by GEMM before being overwritten)
  gcn_gemm_wmma_kernel<256, 256><<<gemm_blocks, 256, 0, stream>>>(hB, W2, hA, N);
  post(hA, b2, hB, 256);
  // Layer 3: 256 -> 128
  gcn_gemm_wmma_kernel<256, 128><<<gemm_blocks, 256, 0, stream>>>(hB, W3, hA, N);
  post(hA, b3, outF, 128);
}
